// WindowAttention_38577396253071
// MI455X (gfx1250) — compile-verified, matrix operand order checked by reading, not run on hardware
//
#include <hip/hip_runtime.h>

// ---------------------------------------------------------------------------
// Fused window attention for MI455X (gfx1250, wave32, WMMA).
// One workgroup (256 threads = 8 waves) per window. Everything lives in LDS.
// ---------------------------------------------------------------------------

typedef __attribute__((ext_vector_type(16))) __bf16 v16bf;
typedef __attribute__((ext_vector_type(8)))  __bf16 v8bf;
typedef __attribute__((ext_vector_type(8)))  float  v8f;
typedef __attribute__((ext_vector_type(4)))  float  v4f;

#define DIM   512
#define NTOK  64
#define NH    16
#define HD    32
#define XS_STRIDE 520   // bf16 elems per row (1040B: 16B aligned, bank-rotating)
#define VT_STRIDE 72    // bf16 elems per row (144B: 16B aligned, bank-rotating)

// LDS layout (elements of __bf16)
#define OFF_XS  0                        // 64*520  (x+pe bf16; reused as attn-out)
#define OFF_QS  (64*XS_STRIDE)
#define OFF_KS  (2*64*XS_STRIDE)
#define OFF_VT  (3*64*XS_STRIDE)         // 512*72, V stored transposed (ch-major)
#define LDS_ELEMS (3*64*XS_STRIDE + 512*VT_STRIDE)   // 136704 elems = 273408 B

static __device__ inline v16bf cat8(v8bf a, v8bf b) {
  return __builtin_shufflevector(a, b, 0,1,2,3,4,5,6,7,8,9,10,11,12,13,14,15);
}
static __device__ inline v8f wmma_bf16(v16bf a, v16bf b, v8f c) {
  return __builtin_amdgcn_wmma_f32_16x16x32_bf16(false, a, false, b, (short)0, c,
                                                 false, false);
}
static __device__ inline v8bf ld8(const __bf16* p) { return *(const v8bf*)p; }

// ---------------------------------------------------------------------------
// One-time f32 -> bf16 weight conversion into d_ws: [Wq | Wk | Wv | Wp]
// ---------------------------------------------------------------------------
__global__ void convert_weights(const float* __restrict__ Wq,
                                const float* __restrict__ Wk,
                                const float* __restrict__ Wv,
                                const float* __restrict__ Wp,
                                __bf16* __restrict__ ws) {
  int i = blockIdx.x * blockDim.x + threadIdx.x;
  const int n = DIM * DIM;
  if (i < n) {
    ws[i]       = (__bf16)Wq[i];
    ws[n + i]   = (__bf16)Wk[i];
    ws[2*n + i] = (__bf16)Wv[i];
    ws[3*n + i] = (__bf16)Wp[i];
  }
}

// ---------------------------------------------------------------------------
// Fused main kernel
// ---------------------------------------------------------------------------
__global__ __launch_bounds__(256)
void window_attn(const float* __restrict__ x, const float* __restrict__ pe,
                 const __bf16* __restrict__ wgt,
                 const float* __restrict__ bq, const float* __restrict__ bk,
                 const float* __restrict__ bv, const float* __restrict__ bp,
                 const float* __restrict__ lw, const float* __restrict__ lb,
                 float* __restrict__ out) {
  extern __shared__ __bf16 smem[];
  __bf16* xs  = smem + OFF_XS;   // x+pe (bf16), later reused as attention output
  __bf16* qs  = smem + OFF_QS;
  __bf16* ksm = smem + OFF_KS;
  __bf16* vt  = smem + OFF_VT;   // V transposed: [channel][token]

  const int tid  = threadIdx.x;
  const int lane = tid & 31;
  const int wave = tid >> 5;
  const int l15  = lane & 15;
  const int hi   = (lane >> 4) & 1;   // lane half (wave32)
  const int koff = hi * 8;            // K sub-offset per A/B fragment layout

  const int b = blockIdx.x;
  const float* xg = x + (size_t)b * NTOK * DIM;

  // ---- Phase 1: async-DMA x tile (f32, 128KB) into LDS staging, then
  //      convert + add abs_pe -> xs (bf16). Staging overlays qs/ks regions.
  {
    __bf16* stage = smem + OFF_QS;
    unsigned sbase = (unsigned)(unsigned long long)(void*)stage;  // LDS byte off
    for (int i = tid; i < (NTOK * DIM) / 4; i += 256) {           // 16B chunks
      unsigned dst = sbase + (unsigned)(i * 16);
      const float* src = xg + i * 4;
      asm volatile("global_load_async_to_lds_b128 %0, %1, off"
                   :: "v"(dst), "v"(src) : "memory");
    }
    asm volatile("s_wait_asynccnt 0" ::: "memory");
    __syncthreads();
    const float* stf = (const float*)stage;
    for (int i = tid; i < NTOK * DIM; i += 256) {
      float v = stf[i] + pe[i];
      xs[(i >> 9) * XS_STRIDE + (i & 511)] = (__bf16)v;
    }
  }
  __syncthreads();

  // ---- Phase 2: Q/K/V projections.  q = x@Wq^T + bq  (M=64,N=512,K=512)
  // 3 projections * 128 (16x16) tiles, 16 k-steps each, over 8 waves.
  for (int job = wave; job < 3 * 128; job += 8) {
    const int proj = job >> 7;     // 0=q 1=k 2=v
    const int t    = job & 127;
    const int mt   = t >> 5;       // token tile 0..3
    const int nt   = t & 31;       // channel tile 0..31
    const __bf16* W = wgt + (size_t)proj * (DIM * DIM);
    v8f acc = {0.f,0.f,0.f,0.f,0.f,0.f,0.f,0.f};
    const int arow = (mt * 16 + l15) * XS_STRIDE;  // A: row of x-tile
    const int brow = (nt * 16 + l15) * DIM;        // B col n = weight row d
    for (int kk = 0; kk < 16; ++kk) {
      const int kb = kk * 32 + koff;
      v16bf a = cat8(ld8(xs + arow + kb), ld8(xs + arow + kb + 16));
      v16bf bb = cat8(ld8(W + brow + kb), ld8(W + brow + kb + 16));
      acc = wmma_bf16(a, bb, acc);
    }
    const float* bias = (proj == 0) ? bq : ((proj == 1) ? bk : bv);
    const float bval = bias[nt * 16 + l15];
    if (proj < 2) {
      __bf16* dst = (proj == 0) ? qs : ksm;
      #pragma unroll
      for (int r = 0; r < 8; ++r) {
        int m = mt * 16 + r + 8 * hi;
        dst[m * XS_STRIDE + nt * 16 + l15] = (__bf16)(acc[r] + bval);
      }
    } else {  // V stored transposed (channel-major) for V^T A-fragments + LePE
      #pragma unroll
      for (int r = 0; r < 8; ++r) {
        int m = mt * 16 + r + 8 * hi;
        vt[(nt * 16 + l15) * VT_STRIDE + m] = (__bf16)(acc[r] + bval);
      }
    }
  }
  __syncthreads();

  // ---- Phase 3: attention, 2 heads per wave.
  // Compute S^T = K @ Q^T so softmax over keys is lane-local + one shfl_xor(16),
  // and P (in C/D layout) feeds the O^T = V^T @ P WMMA as a B-fragment with a
  // purely element-wise per-lane repack (no cross-lane movement).
  const float scale = 0.17677669529663687f;  // 1/sqrt(32)
  __bf16* os = xs;                           // reuse xs region for attn output
  for (int hh = 0; hh < 2; ++hh) {
    const int h  = wave + hh * 8;
    const int hb = h * HD;
    v8f st[4][4];
    #pragma unroll
    for (int mt = 0; mt < 4; ++mt) {         // key tiles (A = K rows)
      const int ar = (mt * 16 + l15) * XS_STRIDE + hb;
      v16bf a = cat8(ld8(ksm + ar + koff), ld8(ksm + ar + koff + 16));
      #pragma unroll
      for (int nt = 0; nt < 4; ++nt) {       // query tiles (B cols = Q rows)
        const int br = (nt * 16 + l15) * XS_STRIDE + hb;
        v16bf bb = cat8(ld8(qs + br + koff), ld8(qs + br + koff + 16));
        v8f z = {0.f,0.f,0.f,0.f,0.f,0.f,0.f,0.f};
        st[mt][nt] = wmma_bf16(a, bb, z);
      }
    }
    // softmax over keys (4 mt-tiles x 8 elems in-lane, + opposite lane half)
    #pragma unroll
    for (int nt = 0; nt < 4; ++nt) {
      float mx = -1e30f;
      #pragma unroll
      for (int mt = 0; mt < 4; ++mt)
        #pragma unroll
        for (int r = 0; r < 8; ++r) mx = fmaxf(mx, st[mt][nt][r]);
      mx = fmaxf(mx, __shfl_xor(mx, 16, 32));
      float sum = 0.f;
      #pragma unroll
      for (int mt = 0; mt < 4; ++mt)
        #pragma unroll
        for (int r = 0; r < 8; ++r) {
          float p = __expf((st[mt][nt][r] - mx) * scale);
          st[mt][nt][r] = p;
          sum += p;
        }
      sum += __shfl_xor(sum, 16, 32);
      const float inv = 1.f / sum;
      #pragma unroll
      for (int mt = 0; mt < 4; ++mt)
        #pragma unroll
        for (int r = 0; r < 8; ++r) st[mt][nt][r] *= inv;
    }
    // O^T = V^T @ P : M=32 (2 ch tiles), N=64 (4 query tiles), K=64 (2 steps)
    #pragma unroll
    for (int mt2 = 0; mt2 < 2; ++mt2) {
      const int ar = (hb + mt2 * 16 + l15) * VT_STRIDE;
      #pragma unroll
      for (int nt = 0; nt < 4; ++nt) {
        v8f o = {0.f,0.f,0.f,0.f,0.f,0.f,0.f,0.f};
        #pragma unroll
        for (int kt = 0; kt < 2; ++kt) {
          const int kb = kt * 32 + koff;
          v16bf a = cat8(ld8(vt + ar + kb), ld8(vt + ar + kb + 16));
          v16bf bb;                             // P as B-frag: element-wise
          #pragma unroll
          for (int e = 0; e < 8; ++e) {
            bb[e]     = (__bf16)st[2 * kt][nt][e];
            bb[e + 8] = (__bf16)st[2 * kt + 1][nt][e];
          }
          o = wmma_bf16(a, bb, o);
        }
        #pragma unroll
        for (int r = 0; r < 8; ++r) {
          int q = nt * 16 + l15;
          int d = hb + mt2 * 16 + r + 8 * hi;
          os[q * XS_STRIDE + d] = (__bf16)o[r];
        }
      }
    }
  }
  __syncthreads();

  // ---- Phase 4: LePE depthwise 3x3 over V's 8x8 grid, accumulate into os.
  for (int i = tid; i < DIM * NTOK; i += 256) {
    const int c = i >> 6;
    const int t = i & 63;
    const int ii = t >> 3, jj = t & 7;
    const __bf16* vrow = vt + c * VT_STRIDE;
    float a = lb[c];
    #pragma unroll
    for (int di = -1; di <= 1; ++di)
      #pragma unroll
      for (int dj = -1; dj <= 1; ++dj) {
        int i2 = ii + di, j2 = jj + dj;
        if (i2 >= 0 && i2 < 8 && j2 >= 0 && j2 < 8)
          a += (float)vrow[i2 * 8 + j2] * lw[c * 9 + (di + 1) * 3 + (dj + 1)];
      }
    os[t * XS_STRIDE + c] = (__bf16)((float)os[t * XS_STRIDE + c] + a);
  }
  __syncthreads();

  // ---- Phase 5: output projection  out = os @ Wp^T + bp  (f32 to global)
  const __bf16* Wp = wgt + 3 * (size_t)(DIM * DIM);
  float* og = out + (size_t)b * NTOK * DIM;
  for (int t = wave; t < 128; t += 8) {
    const int mt = t >> 5;
    const int nt = t & 31;
    v8f acc = {0.f,0.f,0.f,0.f,0.f,0.f,0.f,0.f};
    const int arow = (mt * 16 + l15) * XS_STRIDE;
    const int brow = (nt * 16 + l15) * DIM;
    for (int kk = 0; kk < 16; ++kk) {
      const int kb = kk * 32 + koff;
      v16bf a = cat8(ld8(os + arow + kb), ld8(os + arow + kb + 16));
      v16bf bb = cat8(ld8(Wp + brow + kb), ld8(Wp + brow + kb + 16));
      acc = wmma_bf16(a, bb, acc);
    }
    const float bval = bp[nt * 16 + l15];
    #pragma unroll
    for (int r = 0; r < 8; ++r) {
      int m = mt * 16 + r + 8 * hi;
      og[m * DIM + nt * 16 + l15] = acc[r] + bval;
    }
  }
}

// ---------------------------------------------------------------------------
extern "C" void kernel_launch(void* const* d_in, const int* in_sizes, int n_in,
                              void* d_out, int out_size, void* d_ws, size_t ws_size,
                              hipStream_t stream) {
  const float* x  = (const float*)d_in[0];
  const float* pe = (const float*)d_in[1];
  const float* Wq = (const float*)d_in[2];
  const float* bq = (const float*)d_in[3];
  const float* Wk = (const float*)d_in[4];
  const float* bk = (const float*)d_in[5];
  const float* Wv = (const float*)d_in[6];
  const float* bv = (const float*)d_in[7];
  const float* lw = (const float*)d_in[8];
  const float* lb = (const float*)d_in[9];
  const float* Wp = (const float*)d_in[10];
  const float* bp = (const float*)d_in[11];
  float* out = (float*)d_out;
  __bf16* wgt = (__bf16*)d_ws;   // needs 4*512*512*2 = 2 MB scratch

  convert_weights<<<(DIM * DIM + 255) / 256, 256, 0, stream>>>(Wq, Wk, Wv, Wp, wgt);

  const size_t smem = (size_t)LDS_ELEMS * sizeof(__bf16);  // 273408 B
  hipFuncSetAttribute((const void*)window_attn,
                      hipFuncAttributeMaxDynamicSharedMemorySize, (int)smem);
  window_attn<<<2048, 256, smem, stream>>>(x, pe, wgt, bq, bk, bv, bp, lw, lb, out);
}
